// GIN_32796370273146
// MI455X (gfx1250) — matrix operand order repaired and checked by reading, not compiled, not run on hardware
//
#include <hip/hip_runtime.h>

#define HH   256
#define NND  50000
#define EED  800000
#define LLD  5
#define GGD  128
#define BNEPS 1e-5f

#define KCH    128          // K chunk staged in LDS
#define APITCH 136          // LDS row pitch in halves (272B = 68 DW -> 4-bank lane stride)
#define BPITCH 136

typedef __attribute__((ext_vector_type(16))) _Float16 v16h;
typedef __attribute__((ext_vector_type(8)))  _Float16 v8h;
typedef __attribute__((ext_vector_type(8)))  float    v8f;

// ---------------------------------------------------------------- helpers

// A-fragment (16x32 f16, documented layout): lane holds row m=lane&15,
// kb=(lane>>4)*8; halves j0..7 -> K=kb+0..7, j8..15 -> K=kb+16..23.
__device__ __forceinline__ v16h load_frag_a(const _Float16* p) {
  v8h lo = *(const v8h*)(p);
  v8h hi = *(const v8h*)(p + 16);
  v16h r;
#pragma unroll
  for (int j = 0; j < 8; ++j) { r[j] = lo[j]; r[j + 8] = hi[j]; }
  return r;
}

// B-fragment (32x16 f16): lane holds col n=lane&15, kb=(lane>>4)*8; K=kb+0..7, kb+16..23.
__device__ __forceinline__ v16h load_frag_b(const _Float16* p) {
  v8h lo = *(const v8h*)(p);
  v8h hi = *(const v8h*)(p + 16);
  v16h r;
#pragma unroll
  for (int j = 0; j < 8; ++j) { r[j] = lo[j]; r[j + 8] = hi[j]; }
  return r;
}

__device__ __forceinline__ v8f wmma_f16(v16h a, v16h b, v8f c) {
  return __builtin_amdgcn_wmma_f32_16x16x32_f16(
      /*neg_a=*/false, a, /*neg_b=*/false, b,
      /*c_mod=*/(short)0, c, /*reuse_a=*/false, /*reuse_b=*/false);
}

// ---------------------------------------------------------------- kernels

// x[n,c] = sum_f atom_emb[f, x_feat[n,f], c]
__global__ void atom_encode_kernel(const int* __restrict__ xf,
                                   const float* __restrict__ emb,
                                   float* __restrict__ x) {
  const int n = blockIdx.x, c = threadIdx.x;
  float s = 0.f;
#pragma unroll
  for (int f = 0; f < 9; ++f) {
    const int idx = xf[n * 9 + f];
    s += emb[((size_t)(f * 64 + idx)) * HH + c];
  }
  x[(size_t)n * HH + c] = s;
}

__global__ void copy_f32_kernel(const float* __restrict__ src,
                                float* __restrict__ dst, int n) {
  const int i = blockIdx.x * blockDim.x + threadIdx.x;
  if (i < n) dst[i] = src[i];
}

__global__ void zero_f32_kernel(float* __restrict__ dst, int n) {
  const int i = blockIdx.x * blockDim.x + threadIdx.x;
  if (i < n) dst[i] = 0.f;
}

__global__ void cvt_f16_kernel(const float* __restrict__ src,
                               _Float16* __restrict__ dst, int n) {
  const int i = blockIdx.x * blockDim.x + threadIdx.x;
  if (i < n) dst[i] = (_Float16)src[i];
}

// wt[n,k] = (f16) w[k,n]   (transposed so B-fragments are contiguous in K)
__global__ void wprep_kernel(const float* __restrict__ w,
                             _Float16* __restrict__ wt) {
  const int n = blockIdx.x, k = threadIdx.x;
  wt[(size_t)n * HH + k] = (_Float16)w[(size_t)k * HH + n];
}

// GINE message: agg[dst] += relu(x[src] + bond_embed(edge)); bond table is
// 24KB -> L2 resident, so e[E,H] is never materialized (saves ~4GB HBM).
__global__ void edge_message_kernel(const int* __restrict__ ei,
                                    const int* __restrict__ ea,
                                    const float* __restrict__ bemb,
                                    const float* __restrict__ x,
                                    float* __restrict__ agg) {
  const int e = blockIdx.x, c = threadIdx.x;
  const int s = ei[e];
  const int d = ei[EED + e];
  float v = x[(size_t)s * HH + c];
#pragma unroll
  for (int f = 0; f < 3; ++f) {
    const int a = ea[e * 3 + f];
    v += bemb[((size_t)(f * 8 + a)) * HH + c];
  }
  v = fmaxf(v, 0.f);
  atomicAdd(agg + (size_t)d * HH + c, v);
}

// C[rows,256] = A[rows,256] @ Bt^T + bias, f16 in / f32 out, WMMA core.
// Block: 8 waves -> 128x64 tile; wave -> 32x32 tile (4 accumulators).
// A/B staged through LDS (coalesced global b128 in, bank-conflict-free
// ds_load_b128 fragment reads via pitch-136 padding).
__global__ __launch_bounds__(256)
void gemm_wmma_kernel(const _Float16* __restrict__ A,
                      const _Float16* __restrict__ Bt,
                      const float* __restrict__ bias,
                      float* __restrict__ C, int rows, int do_relu) {
  __shared__ _Float16 As[128 * APITCH];   // 34,816 B
  __shared__ _Float16 Bs[64 * BPITCH];    // 17,408 B

  const int tid  = threadIdx.x;
  const int wave = tid >> 5;
  const int lane = tid & 31;
  const int wm = (wave >> 1) * 32;        // wave row offset in block tile
  const int wn = (wave & 1) * 32;         // wave col offset in block tile
  const int m0 = blockIdx.x * 128;
  const int n0 = blockIdx.y * 64;
  const int t16 = lane & 15;
  const int hi  = lane >> 4;              // 0 or 1
  const int kba = hi << 3;                // A frag: +0 / +8
  const int kbb = hi << 3;                // B frag: +0 / +8

  // copy-in mapping: 16 threads x 16B cover one 256B (128-half) row segment
  const int crow = tid >> 4;              // 0..15
  const int ccol = (tid & 15) * 8;        // halves: 0..120

  v8f acc00 = {}, acc01 = {}, acc10 = {}, acc11 = {};

  for (int kc = 0; kc < HH; kc += KCH) {
    // stage A: 128 rows x 128 halves (8 passes of 16 rows)
#pragma unroll
    for (int p = 0; p < 8; ++p) {
      const int r = p * 16 + crow;
      int gr = m0 + r; if (gr >= rows) gr = rows - 1;
      *(v8h*)(&As[r * APITCH + ccol]) =
          *(const v8h*)(A + (size_t)gr * HH + kc + ccol);
    }
    // stage B: 64 rows x 128 halves (4 passes of 16 rows)
#pragma unroll
    for (int p = 0; p < 4; ++p) {
      const int r = p * 16 + crow;
      *(v8h*)(&Bs[r * BPITCH + ccol]) =
          *(const v8h*)(Bt + (size_t)(n0 + r) * HH + kc + ccol);
    }
    if (kc + KCH < HH) {                  // prefetch next K chunk (global_prefetch_b8)
      __builtin_prefetch(A + (size_t)(m0 + crow * 8) * HH + kc + KCH, 0, 3);
      __builtin_prefetch(Bt + (size_t)(n0 + (crow & 3) * 16) * HH + kc + KCH, 0, 3);
    }
    __syncthreads();

#pragma unroll
    for (int kt = 0; kt < KCH; kt += 32) {
      const v16h a0 = load_frag_a(&As[(wm + t16)      * APITCH + kt + kba]);
      const v16h a1 = load_frag_a(&As[(wm + 16 + t16) * APITCH + kt + kba]);
      const v16h b0 = load_frag_b(&Bs[(wn + t16)      * BPITCH + kt + kbb]);
      const v16h b1 = load_frag_b(&Bs[(wn + 16 + t16) * BPITCH + kt + kbb]);
      acc00 = wmma_f16(a0, b0, acc00);
      acc01 = wmma_f16(a0, b1, acc01);
      acc10 = wmma_f16(a1, b0, acc10);
      acc11 = wmma_f16(a1, b1, acc11);
    }
    __syncthreads();
  }

  // C/D layout: VGPR i, lane L -> row = i + (L>=16 ? 8 : 0), col = L&15
  const int col0 = n0 + wn + t16;
  const int col1 = col0 + 16;
  const int rbase = m0 + wm + (hi << 3);
  const float bia0 = bias[col0];
  const float bia1 = bias[col1];
#pragma unroll
  for (int i = 0; i < 8; ++i) {
    const int r0 = rbase + i;
    const int r1 = r0 + 16;
    if (r0 < rows) {
      float v = acc00[i] + bia0; if (do_relu) v = fmaxf(v, 0.f);
      C[(size_t)r0 * HH + col0] = v;
      v = acc01[i] + bia1; if (do_relu) v = fmaxf(v, 0.f);
      C[(size_t)r0 * HH + col1] = v;
    }
    if (r1 < rows) {
      float v = acc10[i] + bia0; if (do_relu) v = fmaxf(v, 0.f);
      C[(size_t)r1 * HH + col0] = v;
      v = acc11[i] + bia1; if (do_relu) v = fmaxf(v, 0.f);
      C[(size_t)r1 * HH + col1] = v;
    }
  }
}

// Per-column sum / sum-of-squares (coalesced: thread == column), atomics into
// stats[0..H) = sum, stats[H..2H) = sumsq.
__global__ void bn_stats_kernel(const float* __restrict__ y,
                                float* __restrict__ stats, int rows) {
  const int c = threadIdx.x;
  const int r0 = blockIdx.x * 256;
  int rend = r0 + 256; if (rend > rows) rend = rows;
  float s = 0.f, s2 = 0.f;
  for (int r = r0; r < rend; ++r) {
    const float v = y[(size_t)r * HH + c];
    s += v; s2 += v * v;
  }
  atomicAdd(stats + c, s);
  atomicAdd(stats + HH + c, s2);
}

__global__ void bn_apply_kernel(const float* __restrict__ y,
                                const float* __restrict__ stats,
                                const float* __restrict__ g,
                                const float* __restrict__ be,
                                _Float16* __restrict__ d16,
                                float* __restrict__ d32,
                                int n, int rows, int do_relu) {
  const int i = blockIdx.x * blockDim.x + threadIdx.x;
  if (i >= n) return;
  const int c = i & (HH - 1);
  const float inv = 1.f / (float)rows;
  const float mu = stats[c] * inv;
  const float var = stats[HH + c] * inv - mu * mu;
  float v = (y[i] - mu) * rsqrtf(var + BNEPS) * g[c] + be[c];
  if (do_relu) v = fmaxf(v, 0.f);
  if (d16) d16[i] = (_Float16)v;
  if (d32) d32[i] = v;
}

// global_add_pool: out[batch[n], c] += t[n, c]
__global__ void pool_kernel(const float* __restrict__ t,
                            const int* __restrict__ batch,
                            float* __restrict__ out) {
  const int n = blockIdx.x, c = threadIdx.x;
  atomicAdd(out + (size_t)batch[n] * HH + c, t[(size_t)n * HH + c]);
}

// ---------------------------------------------------------------- driver

extern "C" void kernel_launch(void* const* d_in, const int* in_sizes, int n_in,
                              void* d_out, int out_size, void* d_ws, size_t ws_size,
                              hipStream_t stream) {
  (void)in_sizes; (void)n_in; (void)out_size; (void)ws_size;

  const int*   x_feat   = (const int*)d_in[0];
  const int*   ei       = (const int*)d_in[1];
  const int*   ea       = (const int*)d_in[2];
  const int*   batch    = (const int*)d_in[3];
  const float* atom_emb = (const float*)d_in[4];
  const float* bond_emb = (const float*)d_in[5];
  const float* W1  = (const float*)d_in[6];
  const float* b1  = (const float*)d_in[7];
  const float* g1  = (const float*)d_in[8];
  const float* be1 = (const float*)d_in[9];
  const float* W2  = (const float*)d_in[10];
  const float* b2  = (const float*)d_in[11];
  const float* g2  = (const float*)d_in[12];
  const float* be2 = (const float*)d_in[13];
  const float* outW1 = (const float*)d_in[14];
  const float* outb1 = (const float*)d_in[15];
  const float* outg  = (const float*)d_in[16];
  const float* outbe = (const float*)d_in[17];
  const float* outW2 = (const float*)d_in[18];
  const float* outb2 = (const float*)d_in[19];
  float* out = (float*)d_out;

  const size_t NHn = (size_t)NND * HH;
  const size_t HHs = (size_t)HH * HH;

  // workspace carve-out (256B aligned)
  char*  ws  = (char*)d_ws;
  size_t off = 0;
  auto carve = [&](size_t bytes) -> void* {
    void* p = ws + off;
    off = (off + bytes + 255) & ~(size_t)255;
    return p;
  };
  float*     xbuf  = (float*)carve(NHn * 4);
  float*     agg   = (float*)carve(NHn * 4);
  float*     ybuf  = (float*)carve(NHn * 4);
  _Float16*  h16   = (_Float16*)carve(NHn * 2);
  _Float16*  y16   = (_Float16*)carve(NHn * 2);
  float*     stats = (float*)carve(2 * HH * 4);
  _Float16*  wt16  = (_Float16*)carve(12 * HHs * 2);

  const int flatB = 256;
  const int flatG = (int)((NHn + flatB - 1) / flatB);
  const dim3 gemmG((NND + 127) / 128, HH / 64);

  // --- weight prep: transposed f16 copies ---
  for (int l = 0; l < LLD; ++l) {
    wprep_kernel<<<HH, HH, 0, stream>>>(W1 + l * HHs, wt16 + l * HHs);
    wprep_kernel<<<HH, HH, 0, stream>>>(W2 + l * HHs, wt16 + (LLD + l) * HHs);
  }
  wprep_kernel<<<HH, HH, 0, stream>>>(outW1, wt16 + 10 * HHs);
  wprep_kernel<<<HH, HH, 0, stream>>>(outW2, wt16 + 11 * HHs);

  // --- atom encoder ---
  atom_encode_kernel<<<NND, HH, 0, stream>>>(x_feat, atom_emb, xbuf);

  // --- GINE layers ---
  for (int l = 0; l < LLD; ++l) {
    copy_f32_kernel<<<flatG, flatB, 0, stream>>>(xbuf, agg, (int)NHn);
    edge_message_kernel<<<EED, HH, 0, stream>>>(ei, ea, bond_emb, xbuf, agg);
    cvt_f16_kernel<<<flatG, flatB, 0, stream>>>(agg, h16, (int)NHn);

    gemm_wmma_kernel<<<gemmG, 256, 0, stream>>>(
        h16, wt16 + l * HHs, b1 + l * HH, ybuf, NND, 0);
    zero_f32_kernel<<<(2 * HH + 255) / 256, 256, 0, stream>>>(stats, 2 * HH);
    bn_stats_kernel<<<(NND + 255) / 256, HH, 0, stream>>>(ybuf, stats, NND);
    bn_apply_kernel<<<flatG, flatB, 0, stream>>>(
        ybuf, stats, g1 + l * HH, be1 + l * HH,
        y16, (float*)nullptr, (int)NHn, NND, 1);

    gemm_wmma_kernel<<<gemmG, 256, 0, stream>>>(
        y16, wt16 + (LLD + l) * HHs, b2 + l * HH, ybuf, NND, 1);
    zero_f32_kernel<<<(2 * HH + 255) / 256, 256, 0, stream>>>(stats, 2 * HH);
    bn_stats_kernel<<<(NND + 255) / 256, HH, 0, stream>>>(ybuf, stats, NND);
    bn_apply_kernel<<<flatG, flatB, 0, stream>>>(
        ybuf, stats, g2 + l * HH, be2 + l * HH,
        (_Float16*)nullptr, xbuf, (int)NHn, NND, 1);
  }

  // --- output head ---
  cvt_f16_kernel<<<flatG, flatB, 0, stream>>>(xbuf, h16, (int)NHn);
  gemm_wmma_kernel<<<gemmG, 256, 0, stream>>>(
      h16, wt16 + 10 * HHs, outb1, ybuf, NND, 0);
  zero_f32_kernel<<<(2 * HH + 255) / 256, 256, 0, stream>>>(stats, 2 * HH);
  bn_stats_kernel<<<(NND + 255) / 256, HH, 0, stream>>>(ybuf, stats, NND);
  bn_apply_kernel<<<flatG, flatB, 0, stream>>>(
      ybuf, stats, outg, outbe, y16, (float*)nullptr, (int)NHn, NND, 1);
  gemm_wmma_kernel<<<gemmG, 256, 0, stream>>>(
      y16, wt16 + 11 * HHs, outb2, ybuf, NND, 0);

  // --- global add pool ---
  zero_f32_kernel<<<(GGD * HH + 255) / 256, 256, 0, stream>>>(out, GGD * HH);
  pool_kernel<<<NND, HH, 0, stream>>>(ybuf, batch, out);
}
